// MLP_Diag_69320772157908
// MI455X (gfx1250) — matrix operand order repaired and checked by reading, not compiled, not run on hardware
//
#include <hip/hip_runtime.h>

#define N_ROWS   10000
#define ISIZE    256
#define KP1      31          // top_k(sim, K+1), K=30
#define NPAD     10048       // 10000 rounded up to multiple of 64 (157*64)
#define MT       157         // macro tiles per dim (64x64 tiles)
#define NTILES   (MT * MT)

typedef _Float16 v8h  __attribute__((ext_vector_type(8)));
typedef _Float16 v16h __attribute__((ext_vector_type(16)));
typedef float    v8f  __attribute__((ext_vector_type(8)));

// ---------------------------------------------------------------------------
// Kernel 1: emb[row] = normalize(relu(features*w0)*w1), stored as f16,
// rows [10000,10048) zero-padded so the GEMM never loads out of bounds.
// ---------------------------------------------------------------------------
__global__ __launch_bounds__(256) void embed_kernel(const float* __restrict__ features,
                                                    const float* __restrict__ w0,
                                                    const float* __restrict__ w1,
                                                    _Float16* __restrict__ emb) {
    __shared__ float red[256];
    const int row = blockIdx.x;
    const int tid = threadIdx.x;

    float h = 0.0f;
    if (row < N_ROWS) {
        float f = features[(size_t)row * ISIZE + tid];
        h = fmaxf(f * w0[tid], 0.0f) * w1[tid];
    }
    red[tid] = h * h;
    __syncthreads();
    #pragma unroll
    for (int s = 128; s > 0; s >>= 1) {
        if (tid < s) red[tid] += red[tid + s];
        __syncthreads();
    }
    float norm  = sqrtf(red[0]);
    float scale = 1.0f / fmaxf(norm, 1e-12f);
    emb[(size_t)row * ISIZE + tid] = (_Float16)(h * scale);
}

// ---------------------------------------------------------------------------
// Kernel 2: sim = emb @ emb^T via v_wmma_f32_16x16x32_f16.
// One wave computes a 64x64 output block (4x4 WMMA tiles), f32 accumulate.
// Per k-step: 8 fragment loads feed 16 WMMAs (2 wmma per fragment load).
// ---------------------------------------------------------------------------
__device__ __forceinline__ v16h cat16(v8h lo, v8h hi) {
    return __builtin_shufflevector(lo, hi, 0, 1, 2, 3, 4, 5, 6, 7,
                                           8, 9, 10, 11, 12, 13, 14, 15);
}

// A fragment (16x32 f16): lane half h -> K in {k+h*8 .. +7} and {k+16+h*8 .. +7}
__device__ __forceinline__ v16h load_afrag(const _Float16* __restrict__ rowp, int k, int h) {
    v8h lo = *(const v8h*)(rowp + k + h * 8);
    v8h hi = *(const v8h*)(rowp + k + 16 + h * 8);
    return cat16(lo, hi);
}

// B fragment (32x16 f16): lane half h -> K in {k+h*8 .. +7} and {k+16+h*8 .. +7}
__device__ __forceinline__ v16h load_bfrag(const _Float16* __restrict__ rowp, int k, int h) {
    v8h lo = *(const v8h*)(rowp + k + h * 8);
    v8h hi = *(const v8h*)(rowp + k + 16 + h * 8);
    return cat16(lo, hi);
}

__device__ __forceinline__ void store_ctile(float* __restrict__ out, v8f c,
                                            int ri, int cj, int n, int h) {
    const int col = cj + n;
    if (col >= N_ROWS) return;
    #pragma unroll
    for (int r = 0; r < 8; ++r) {
        int row = ri + r + 8 * h;              // C layout: VGPR r = M=r (lanes 0-15), M=r+8 (16-31)
        if (row < N_ROWS) out[(size_t)row * N_ROWS + col] = c[r];
    }
}

__global__ __launch_bounds__(256) void gemm_kernel(const _Float16* __restrict__ emb,
                                                   float* __restrict__ out) {
    const int tid    = threadIdx.x;
    const int wave   = tid >> 5;
    const int lane   = tid & 31;
    const int n      = lane & 15;              // M index (A) / N index (B,C)
    const int h      = lane >> 4;              // lane half -> K split
    const int tileId = blockIdx.x * 8 + wave;  // 8 waves per block
    if (tileId >= NTILES) return;              // uniform per-wave exit

    const int ti = tileId / MT;
    const int tj = tileId % MT;
    const int i0 = ti * 64;
    const int j0 = tj * 64;

    const _Float16* rowA[4];
    const _Float16* rowB[4];
    #pragma unroll
    for (int r = 0; r < 4; ++r) {
        rowA[r] = emb + (size_t)(i0 + 16 * r + n) * ISIZE;
        rowB[r] = emb + (size_t)(j0 + 16 * r + n) * ISIZE;  // B col n = emb row j0+16r+n
    }

    v8f acc[4][4];
    #pragma unroll
    for (int r = 0; r < 4; ++r)
        #pragma unroll
        for (int c = 0; c < 4; ++c)
            acc[r][c] = (v8f){};

    #pragma unroll
    for (int kt = 0; kt < ISIZE / 32; ++kt) {
        const int k = kt * 32;
        v16h a[4], b[4];
        #pragma unroll
        for (int r = 0; r < 4; ++r) a[r] = load_afrag(rowA[r], k, h);
        #pragma unroll
        for (int c = 0; c < 4; ++c) b[c] = load_bfrag(rowB[c], k, h);
        #pragma unroll
        for (int r = 0; r < 4; ++r)
            #pragma unroll
            for (int c = 0; c < 4; ++c)
                acc[r][c] = __builtin_amdgcn_wmma_f32_16x16x32_f16(
                    false, a[r], false, b[c], (short)0, acc[r][c], false, false);
    }

    #pragma unroll
    for (int r = 0; r < 4; ++r)
        #pragma unroll
        for (int c = 0; c < 4; ++c)
            store_ctile(out, acc[r][c], i0 + 16 * r, j0 + 16 * c, n, h);
}

// ---------------------------------------------------------------------------
// Kernel 3: per-row top-31 threshold + mask + relu, in place on d_out.
// Row cached in LDS as sortable-uint keys; 32-step bitwise binary search
// finds the largest threshold t with count(key >= t) >= 31 (== 31st largest).
// ---------------------------------------------------------------------------
__global__ __launch_bounds__(256) void topk_kernel(float* __restrict__ sim) {
    __shared__ unsigned skeys[N_ROWS];   // 40 KB of the 320 KB/WGP LDS
    __shared__ int scnt;

    const int row = blockIdx.x;
    const int tid = threadIdx.x;
    float* rowp = sim + (size_t)row * N_ROWS;

    // load row, convert to order-preserving unsigned keys
    for (int idx = tid; idx < N_ROWS; idx += 256) {
        unsigned u = __float_as_uint(rowp[idx]);
        skeys[idx] = (u & 0x80000000u) ? ~u : (u | 0x80000000u);
    }
    if (tid == 0) scnt = 0;
    __syncthreads();

    unsigned t = 0u;
    #pragma unroll 1
    for (int b = 31; b >= 0; --b) {
        unsigned cand = t | (1u << b);
        int local = 0;
        for (int idx = tid; idx < N_ROWS; idx += 256)
            local += (skeys[idx] >= cand) ? 1 : 0;
        atomicAdd(&scnt, local);
        __syncthreads();
        if (scnt >= KP1) t = cand;           // scnt uniform across block
        __syncthreads();
        if (tid == 0) scnt = 0;
        __syncthreads();
    }

    // keep only keys >= t; relu: kept negatives -> 0. key MSB==1 <=> value >= +0
    for (int idx = tid; idx < N_ROWS; idx += 256) {
        unsigned k = skeys[idx];
        float v = 0.0f;
        if (k >= t && (k & 0x80000000u))
            v = __uint_as_float(k & 0x7FFFFFFFu);
        rowp[idx] = v;
    }
}

// ---------------------------------------------------------------------------
extern "C" void kernel_launch(void* const* d_in, const int* in_sizes, int n_in,
                              void* d_out, int out_size, void* d_ws, size_t ws_size,
                              hipStream_t stream) {
    const float* features = (const float*)d_in[0];
    const float* w0       = (const float*)d_in[1];
    const float* w1       = (const float*)d_in[2];
    float*       out      = (float*)d_out;
    _Float16*    emb      = (_Float16*)d_ws;   // NPAD * ISIZE * 2B = ~5.14 MB

    embed_kernel<<<NPAD, 256, 0, stream>>>(features, w0, w1, emb);
    gemm_kernel<<<(NTILES + 7) / 8, 256, 0, stream>>>(emb, out);
    topk_kernel<<<N_ROWS, 256, 0, stream>>>(out);
}
